// QuantumClassifier_v2_24756191494221
// MI455X (gfx1250) — matrix operand order repaired and checked by reading, not compiled, not run on hardware
//
#include <hip/hip_runtime.h>
#include <hip/hip_bf16.h>
#include <math.h>

// ---------------------------------------------------------------------------
// Quantum photonic classifier forward pass for MI455X (gfx1250, wave32).
//
// Pipeline (all on `stream`, graph-capture safe, scratch only in d_ws):
//   1. build_unitaries : 4 x (45-MZI Clements mesh) -> 10x10 complex U's
//   2. wmma_gemm_bias  : h = (x @ W_down + b_down) / pi        [WMMA f16->f32]
//   3. diag_unitary    : U1[b] = WR1 diag(e^{ih}) WL1
//   4. permanents      : e1[b,k] = |perm(5x5)|^2   (252 no-bunching rows)
//   5. phi2_kernel     : phi2[b,m] = sum_{k%10==m} e1[b,k]
//   6. diag_unitary    : UF[b] = WRF diag(e^{i phi2}) WLF
//   7. permanents      : probs[b,k] = |perm|^2 / norm (2002 bunched rows)
//   8. wmma_gemm_bias  : out = probs @ W_out + b_out           [WMMA f16->f32]
// ---------------------------------------------------------------------------

typedef __attribute__((ext_vector_type(16))) _Float16 v16h;
typedef __attribute__((ext_vector_type(8)))  float    v8f;

struct C2 { float r, i; };
__device__ __forceinline__ C2 cmul(C2 a, C2 b) { return {a.r*b.r - a.i*b.i, a.r*b.i + a.i*b.r}; }
__device__ __forceinline__ C2 cadd(C2 a, C2 b) { return {a.r + b.r, a.i + b.i}; }
__device__ __forceinline__ C2 cscale(C2 a, float s) { return {a.r*s, a.i*s}; }

// ---------------------------------------------------------------------------
// Kernel 1: build the four 10x10 unitaries from MZI phases.
// One block per unitary; lane j owns column j of U (kept in registers).
// 45 MZIs fully unrolled so row indices are compile-time constants.
// ---------------------------------------------------------------------------
__global__ void build_unitaries(const float* __restrict__ pl1, const float* __restrict__ pr1,
                                const float* __restrict__ plf, const float* __restrict__ prf,
                                float* __restrict__ U4) {
  const float* ph = (blockIdx.x == 0) ? pl1 : (blockIdx.x == 1) ? pr1
                  : (blockIdx.x == 2) ? plf : prf;
  float* outp = U4 + (size_t)blockIdx.x * 200;
  const int j = threadIdx.x;

  float ur[10], ui[10];
#pragma unroll
  for (int i = 0; i < 10; ++i) { ur[i] = (i == j) ? 1.f : 0.f; ui[i] = 0.f; }

  const float s = 0.70710678118654752f;
  int k = 0;
#pragma unroll
  for (int layer = 0; layer < 10; ++layer) {
#pragma unroll
    for (int p = (layer & 1); p < 9; p += 2) {
      float t1 = ph[2 * k], t2 = ph[2 * k + 1];
      ++k;
      float s1, c1, s2c, c2c;
      __sincosf(t1, &s1, &c1);
      __sincosf(t2, &s2c, &c2c);
      C2 e1c{c1, s1}, e2c{c2c, s2c};
      C2 is{0.f, s};
      // M1 = PS(t1) @ B  (B = s*[[1,i],[i,1]])
      C2 M100 = cscale(e1c, s);
      C2 M101 = cmul(e1c, is);
      C2 M110 = is;
      C2 M111 = {s, 0.f};
      // M2 = B @ M1
      C2 M200 = cadd(cscale(M100, s), cmul(is, M110));
      C2 M201 = cadd(cscale(M101, s), cmul(is, M111));
      C2 M210 = cadd(cmul(is, M100), cscale(M110, s));
      C2 M211 = cadd(cmul(is, M101), cscale(M111, s));
      // T = PS(t2) @ M2
      C2 T00 = cmul(e2c, M200), T01 = cmul(e2c, M201);
      C2 T10 = M210, T11 = M211;

      float apr = ur[p],     api = ui[p];
      float bpr = ur[p + 1], bpi = ui[p + 1];
      ur[p]     = T00.r*apr - T00.i*api + T01.r*bpr - T01.i*bpi;
      ui[p]     = T00.r*api + T00.i*apr + T01.r*bpi + T01.i*bpr;
      ur[p + 1] = T10.r*apr - T10.i*api + T11.r*bpr - T11.i*bpi;
      ui[p + 1] = T10.r*api + T10.i*apr + T11.r*bpi + T11.i*bpr;
    }
  }
  if (j < 10) {
#pragma unroll
    for (int i = 0; i < 10; ++i) {
      outp[(i * 10 + j) * 2]     = ur[i];
      outp[(i * 10 + j) * 2 + 1] = ui[i];
    }
  }
}

// ---------------------------------------------------------------------------
// Kernels 2/8: Cout[128 x N] = (A[128 x K] @ B[K x N] + bias) * scale
// One wave per 16-row tile, v_wmma_f32_16x16x32_f16, K chunked by 32.
//
// Fragment mappings (from CDNA5 ISA 7.12.2, wave32):
//   A (16x32 f16):  lane l, row m=l&15, half=l>>4:
//                   a[i] = A_row[k0 + (i<8 ? half*8+i : 16+half*8+(i-8))]
//   B (32x16 f16):  lane l, col n=l&15:  bf[i] = B[(k0 + (i<8 ? half*8+i : 16+half*8+(i-8)))*N + n]
//   C/D (16x16 f32): lane l col n=l&15, VGPR r -> row half*8+r
//
// Full 32-tiles have no bounds guards (branch-free, coalescible loads);
// columns >= N use a clamped address (their outputs are never stored).
// ---------------------------------------------------------------------------
__device__ __forceinline__ v8f gemm_tile32(const float* __restrict__ arow,
                                           const float* __restrict__ B,
                                           int N, int ncl, int k0, int half, v8f acc) {
  float av[16], bv[16];
#pragma unroll
  for (int t = 0; t < 8; ++t) av[t] = arow[k0 + half * 8 + t];
#pragma unroll
  for (int t = 0; t < 8; ++t) av[8 + t] = arow[k0 + 16 + half * 8 + t];
#pragma unroll
  for (int t = 0; t < 16; ++t) bv[t] = B[(size_t)(k0 + ((t < 8) ? (half * 8 + t) : (16 + half * 8 + (t - 8)))) * N + ncl];
  v16h a, bf;
#pragma unroll
  for (int t = 0; t < 16; ++t) {
    a[t]  = (_Float16)av[t];
    bf[t] = (_Float16)bv[t];
  }
  return __builtin_amdgcn_wmma_f32_16x16x32_f16(false, a, false, bf, (short)0, acc, false, false);
}

__global__ void wmma_gemm_bias(const float* __restrict__ A, const float* __restrict__ B,
                               const float* __restrict__ bias, float* __restrict__ Cout,
                               int K, int N, float scale) {
  const int m0   = blockIdx.x * 16;
  const int lane = threadIdx.x & 31;
  const int half = lane >> 4;   // which 16-lane half
  const int mrow = lane & 15;   // A row inside tile
  const int ncol = lane & 15;   // B/C column
  const int ncl  = (ncol < N) ? ncol : 0;  // clamped address for cols >= N

  const float* arow = A + (size_t)(m0 + mrow) * K;
  v8f acc = {};

  const int K64   = K & ~63;
  const int Kfull = K & ~31;
  int k0 = 0;
  for (; k0 < K64; k0 += 64) {  // two WMMAs per iteration for load/compute overlap
    __builtin_prefetch(arow + k0 + 64, 0, 0);  // global_prefetch_b8
    acc = gemm_tile32(arow, B, N, ncl, k0,      half, acc);
    acc = gemm_tile32(arow, B, N, ncl, k0 + 32, half, acc);
  }
  if (k0 < Kfull) {
    acc = gemm_tile32(arow, B, N, ncl, k0, half, acc);
    k0 += 32;
  }
  if (k0 < K) {  // guarded partial tail (runs at most once)
    v16h a, bf;
#pragma unroll
    for (int t = 0; t < 16; ++t) {
      int ka = k0 + ((t < 8) ? (half * 8 + t) : (16 + half * 8 + (t - 8)));
      int kb = ka;
      a[t]  = (ka < K) ? (_Float16)arow[ka] : (_Float16)0.f;
      bf[t] = (kb < K) ? (_Float16)B[(size_t)kb * N + ncl] : (_Float16)0.f;
    }
    acc = __builtin_amdgcn_wmma_f32_16x16x32_f16(false, a, false, bf, (short)0, acc, false, false);
  }

  if (ncol < N) {
#pragma unroll
    for (int r = 0; r < 8; ++r) {
      int m = half * 8 + r;
      Cout[(size_t)(m0 + m) * N + ncol] = (acc[r] + bias[ncol]) * scale;
    }
  }
}

// ---------------------------------------------------------------------------
// Kernels 3/6: Uout[b,m,n] = sum_k Wr[m,k] * e^{i theta[b,k]} * Wl[k,n]
// One block per batch element; threads 0..99 own (m,n).
// ---------------------------------------------------------------------------
__global__ void diag_unitary(const float* __restrict__ Wr, const float* __restrict__ Wl,
                             const float* __restrict__ theta, float* __restrict__ Uout) {
  const int b = blockIdx.x;
  const int t = threadIdx.x;
  __shared__ float pr[10], pi[10];
  if (t < 10) {
    float th = theta[b * 10 + t];
    pr[t] = cosf(th);
    pi[t] = sinf(th);
  }
  __syncthreads();
  if (t < 100) {
    const int m = t / 10, n = t % 10;
    float accr = 0.f, acci = 0.f;
#pragma unroll
    for (int k = 0; k < 10; ++k) {
      float wr = Wr[(m * 10 + k) * 2], wi = Wr[(m * 10 + k) * 2 + 1];
      float ar = wr * pr[k] - wi * pi[k];
      float ai = wr * pi[k] + wi * pr[k];
      float lr = Wl[(k * 10 + n) * 2], li = Wl[(k * 10 + n) * 2 + 1];
      accr += ar * lr - ai * li;
      acci += ar * li + ai * lr;
    }
    Uout[(size_t)b * 200 + (m * 10 + n) * 2]     = accr;
    Uout[(size_t)b * 200 + (m * 10 + n) * 2 + 1] = acci;
  }
}

// ---------------------------------------------------------------------------
// Kernels 4/7: Glynn permanents of 5x5 complex submatrices, one per lane.
// Row sets unranked on the fly (lexicographic combinations; bunched case via
// the combinations(14,5) bijection). Gray-code rowsum updates keep all state
// in registers (loops fully unrolled -> no dynamic register indexing).
// ---------------------------------------------------------------------------
__device__ __forceinline__ int binom_small(int n, int r) {
  if (r < 0 || r > n) return 0;
  int num = 1, den = 1;
  for (int t = 0; t < r; ++t) { num *= (n - t); den *= (t + 1); }
  return num / den;
}

__global__ void permanents(const float* __restrict__ U, float* __restrict__ outp,
                           int Krows, int bunched) {
  const int idx = blockIdx.x * blockDim.x + threadIdx.x;
  const int total = 128 * Krows;
  if (idx >= total) return;
  const int b = idx / Krows;
  int rank = idx - b * Krows;

  // ---- unrank the output Fock row (5 sorted mode indices) ----
  const int nmodes = bunched ? 14 : 10;  // bunched: combos of 14 then subtract position
  int rows[5];
  {
    int v = 0;
#pragma unroll
    for (int pos = 0; pos < 5; ++pos) {
      for (;;) {
        int cnt = binom_small(nmodes - 1 - v, 4 - pos);
        if (rank < cnt) { rows[pos] = v; ++v; break; }
        rank -= cnt; ++v;
      }
    }
  }
  float norm = 1.f;
  if (bunched) {
#pragma unroll
    for (int i = 0; i < 5; ++i) rows[i] -= i;   // -> combinations_with_replacement
    int run = 1;
#pragma unroll
    for (int i = 1; i < 5; ++i) {
      if (rows[i] == rows[i - 1]) { ++run; norm *= (float)run; }
      else run = 1;                              // prod of factorials of multiplicities
    }
  }

  // ---- gather M[i][j] = U[b, rows[i], IN_MODES[j]=2j] (complex) ----
  const float2* Uc = reinterpret_cast<const float2*>(U) + (size_t)b * 100;
  float mr[5][5], mi[5][5];
#pragma unroll
  for (int i = 0; i < 5; ++i) {
#pragma unroll
    for (int j = 0; j < 5; ++j) {
      float2 v = Uc[rows[i] * 10 + 2 * j];
      mr[i][j] = v.x; mi[i][j] = v.y;
    }
  }

  // ---- Glynn with Gray-code over delta_1..4 (delta_0 fixed +1) ----
  float vr[5], vi[5];
#pragma unroll
  for (int j = 0; j < 5; ++j) {
    vr[j] = mr[0][j] + mr[1][j] + mr[2][j] + mr[3][j] + mr[4][j];
    vi[j] = mi[0][j] + mi[1][j] + mi[2][j] + mi[3][j] + mi[4][j];
  }
  float accr, acci;
  {
    float p_r = vr[0], p_i = vi[0];
#pragma unroll
    for (int j = 1; j < 5; ++j) {
      float nr = p_r * vr[j] - p_i * vi[j];
      float ni = p_r * vi[j] + p_i * vr[j];
      p_r = nr; p_i = ni;
    }
    accr = p_r; acci = p_i;
  }
  unsigned gray = 0u;
  float sign = 1.f;
#pragma unroll
  for (int t = 1; t < 16; ++t) {
    const int bit = __builtin_ctz((unsigned)t);  // constant under unroll
    gray ^= (1u << bit);
    const int i = bit + 1;                       // delta_i flipped
    const float s2 = ((gray >> bit) & 1u) ? -2.f : 2.f;
#pragma unroll
    for (int j = 0; j < 5; ++j) {
      vr[j] += s2 * mr[i][j];
      vi[j] += s2 * mi[i][j];
    }
    sign = -sign;
    float p_r = vr[0], p_i = vi[0];
#pragma unroll
    for (int j = 1; j < 5; ++j) {
      float nr = p_r * vr[j] - p_i * vi[j];
      float ni = p_r * vi[j] + p_i * vr[j];
      p_r = nr; p_i = ni;
    }
    accr += sign * p_r;
    acci += sign * p_i;
  }
  // perm = acc/16 ; prob = |perm|^2 / norm
  outp[idx] = (accr * accr + acci * acci) * (1.0f / 256.0f) / norm;
}

// ---------------------------------------------------------------------------
// Kernel 5: phi2[b,m] = sum_{k<252, k%10==m} e1[b,k]
// ---------------------------------------------------------------------------
__global__ void phi2_kernel(const float* __restrict__ e1, float* __restrict__ phi2) {
  const int b = blockIdx.x, m = threadIdx.x;
  if (m < 10) {
    float s = 0.f;
    for (int kk = m; kk < 252; kk += 10) s += e1[(size_t)b * 252 + kk];
    phi2[b * 10 + m] = s;
  }
}

// ---------------------------------------------------------------------------
extern "C" void kernel_launch(void* const* d_in, const int* in_sizes, int n_in,
                              void* d_out, int out_size, void* d_ws, size_t ws_size,
                              hipStream_t stream) {
  (void)in_sizes; (void)n_in; (void)out_size; (void)ws_size;
  const float* x   = (const float*)d_in[0];   // [128,784]
  const float* Wd  = (const float*)d_in[1];   // [784,10]
  const float* bd  = (const float*)d_in[2];   // [10]
  const float* pl1 = (const float*)d_in[3];   // [45,2]
  const float* pr1 = (const float*)d_in[4];
  const float* plf = (const float*)d_in[5];
  const float* prf = (const float*)d_in[6];
  const float* Wo  = (const float*)d_in[7];   // [2002,10]
  const float* bo  = (const float*)d_in[8];   // [10]
  float* out = (float*)d_out;                 // [128,10]

  // Workspace layout (floats); total 343,072 floats = 1.37 MB.
  float* ws    = (float*)d_ws;
  float* U4    = ws;            // 4 x 10x10 complex interleaved (800)
  float* h     = ws + 800;      // [128,10]
  float* U1    = ws + 2080;     // [128,10,10] complex (25600)
  float* e1    = ws + 27680;    // [128,252]
  float* phi2  = ws + 59936;    // [128,10]
  float* UF    = ws + 61216;    // [128,10,10] complex
  float* probs = ws + 86816;    // [128,2002]

  build_unitaries<<<4, 32, 0, stream>>>(pl1, pr1, plf, prf, U4);
  // h = (x @ W_down + b_down) / pi
  wmma_gemm_bias<<<8, 32, 0, stream>>>(x, Wd, bd, h, 784, 10, 0.318309886183790672f);
  // U1 = WR1 diag(e^{ih}) WL1   (WL1 @ U4+0, WR1 @ U4+200)
  diag_unitary<<<128, 128, 0, stream>>>(U4 + 200, U4 + 0, h, U1);
  // 128*252 = 32256 permanents (no-bunching, norm = 1)
  permanents<<<126, 256, 0, stream>>>(U1, e1, 252, 0);
  phi2_kernel<<<128, 32, 0, stream>>>(e1, phi2);
  // UF = WRF diag(e^{i phi2}) WLF   (WLF @ U4+400, WRF @ U4+600)
  diag_unitary<<<128, 128, 0, stream>>>(U4 + 600, U4 + 400, phi2, UF);
  // 128*2002 = 256256 permanents (bunched, divide by multiset norm)
  permanents<<<1002, 256, 0, stream>>>(UF, probs, 2002, 1);
  // out = probs @ W_out + b_out
  wmma_gemm_bias<<<8, 32, 0, stream>>>(probs, Wo, bo, out, 2002, 10, 1.0f);
}
